// VideoDiTBlock_67199058313907
// MI455X (gfx1250) — matrix operand order repaired and checked by reading, not compiled, not run on hardware
//
#include <hip/hip_runtime.h>
#include <hip/hip_bf16.h>
#include <math.h>

// ---- problem constants (from reference) ----
#define B_   2
#define S_   2048
#define T_   (B_*S_)      // 4096 tokens
#define DIM_ 1024
#define H_   16
#define HD_  64
#define E_   8
#define HID_ 4096
#define CAP_ 1280         // ceil(2*2048*2*1.25/8)

typedef __attribute__((ext_vector_type(16))) __bf16 v16bf;
typedef __attribute__((ext_vector_type(8)))  __bf16 v8bf;
typedef __attribute__((ext_vector_type(8)))  float  v8f;

__device__ __forceinline__ v8f wmma_bf16(v16bf a, v16bf b, v8f c) {
  // D = A(16x32 bf16) * B(32x16 bf16) + C(16x16 f32)
  return __builtin_amdgcn_wmma_f32_16x16x32_bf16(false, a, false, b,
                                                 (short)0, c, false, false);
}

// async global->LDS copy, 16B per lane, tracked by ASYNCcnt
__device__ __forceinline__ void async_b128(unsigned lds_off, const __bf16* g) {
  asm volatile("global_load_async_to_lds_b128 %0, %1, off"
               :: "v"(lds_off), "v"(g) : "memory");
}
__device__ __forceinline__ void wait_async0() {
  asm volatile("s_wait_asynccnt 0" ::: "memory");
}
__device__ __forceinline__ unsigned lds_off_of(const void* p) {
  return (unsigned)(size_t)p;  // shared aperture: low 32 bits = LDS offset
}

__device__ __forceinline__ float gelu_f(float x) {
  return 0.5f * x * (1.0f + tanhf(0.7978845608028654f * (x + 0.044715f * x * x * x)));
}

// ---------------- utility kernels ----------------
__global__ void zero_kernel(float* p, long n) {
  long i = (long)blockIdx.x * blockDim.x + threadIdx.x;
  if (i < n) p[i] = 0.0f;
}

__global__ __launch_bounds__(256)
void layernorm_kernel(const float* __restrict__ x, const float* __restrict__ g,
                      const float* __restrict__ b, float* __restrict__ out) {
  __shared__ float s1[256], s2[256];
  const int t = blockIdx.x;
  const float* xr = x + (size_t)t * DIM_;
  float sum = 0.f, sq = 0.f;
  for (int d = threadIdx.x; d < DIM_; d += 256) {
    float v = xr[d]; sum += v; sq += v * v;
  }
  s1[threadIdx.x] = sum; s2[threadIdx.x] = sq;
  __syncthreads();
  for (int st = 128; st > 0; st >>= 1) {
    if (threadIdx.x < st) {
      s1[threadIdx.x] += s1[threadIdx.x + st];
      s2[threadIdx.x] += s2[threadIdx.x + st];
    }
    __syncthreads();
  }
  float mu  = s1[0] * (1.0f / DIM_);
  float var = s2[0] * (1.0f / DIM_) - mu * mu;
  float inv = rsqrtf(var + 1e-6f);
  for (int d = threadIdx.x; d < DIM_; d += 256)
    out[(size_t)t * DIM_ + d] = (xr[d] - mu) * inv * g[d] + b[d];
}

// ---------------- operand swizzle passes (f32 -> fragment-tiled bf16) -------
// A-operand: [batch] MxK row-major f32 -> tiles (M/128 x K/32), each 4096 elem:
//   [sub8][lane32][j16];  lane=(m&15)+16*(g&1), j0=(g>>1)*8  (ISA A-frag map)
__global__ __launch_bounds__(256)
void swizzleA_kernel(const float* __restrict__ in, __bf16* __restrict__ outb,
                     int M, int K, long long sIn, long total) {
  long idx = (long)blockIdx.x * 256 + threadIdx.x;
  if (idx >= total) return;
  long upm = (long)M * (K >> 3);
  int z  = (int)(idx / upm);
  long r = idx % upm;
  int m  = (int)(r / (K >> 3));
  int kq = (int)(r % (K >> 3));
  int kt = kq >> 2, g = kq & 3;
  const float* src = in + (size_t)z * sIn + (size_t)m * K + kq * 8;
  v8bf v;
  for (int i = 0; i < 8; i++) v[i] = (__bf16)src[i];
  int mt = m >> 7, ml = m & 127;
  size_t off = (size_t)z * M * K + ((size_t)mt * (K >> 5) + kt) * 4096
             + (size_t)((ml >> 4) * 512 + ((ml & 15) + 16 * (g & 1)) * 16 + (g >> 1) * 8);
  *(v8bf*)(outb + off) = v;
}

// B-operand: [batch] KxN row-major f32 -> tiles (N/64 x K/32), each 2048 elem:
//   [sub4][lane32][j16];  lane=(n&15)+16*(kg&1), j0=(kg>>1)*8  (ISA B-frag map)
__global__ __launch_bounds__(256)
void swizzleB_kernel(const float* __restrict__ in, __bf16* __restrict__ outb,
                     int K, int N, long long sIn, long total) {
  long idx = (long)blockIdx.x * 256 + threadIdx.x;
  if (idx >= total) return;
  int n  = (int)(idx % N);
  long r = idx / N;
  int kq = (int)(r % (K >> 3));
  int z  = (int)(r / (K >> 3));
  int kt = kq >> 2, kg = kq & 3;
  const float* src = in + (size_t)z * sIn + (size_t)(kq * 8) * N + n;
  v8bf v;
  for (int i = 0; i < 8; i++) v[i] = (__bf16)src[(size_t)i * N];
  int nt = n >> 6, nl = n & 63;
  size_t off = (size_t)z * K * N + ((size_t)nt * (K >> 5) + kt) * 2048
             + (size_t)((nl >> 4) * 512 + ((nl & 15) + 16 * (kg & 1)) * 16 + (kg >> 1) * 8);
  *(v8bf*)(outb + off) = v;
}

// RoPE in place on the Q slice of qkv
__global__ void rope_q_kernel(float* __restrict__ qkv, const float* __restrict__ cs,
                              const float* __restrict__ sn) {
  long i = (long)blockIdx.x * 256 + threadIdx.x;  // over T*H*32
  if (i >= (long)T_ * H_ * 32) return;
  int hd = (int)(i % 32); i /= 32;
  int h  = (int)(i % H_); i /= H_;
  int t  = (int)i;
  int s  = t % S_;
  float* p = qkv + (size_t)t * 3 * DIM_ + h * HD_;
  float t1 = p[hd], t2 = p[hd + 32];
  p[hd]      = t1 * cs[s * HD_ + hd]      - t2 * sn[s * HD_ + hd];
  p[hd + 32] = t2 * cs[s * HD_ + hd + 32] + t1 * sn[s * HD_ + hd + 32];
}

// RoPE(K) fused with swizzle into K^T fragment tiles: per (b,h,keyblock) 2048
// elems: [kk2][nh2][lane32][j16]; lane=(key&15)+16*(hg&1), j0=((hg>>1)&1)*8
__global__ __launch_bounds__(256)
void ropek_swz_kernel(const float* __restrict__ qkv, const float* __restrict__ cs,
                      const float* __restrict__ sn, __bf16* __restrict__ ksz) {
  long idx = (long)blockIdx.x * 256 + threadIdx.x;
  if (idx >= (long)B_ * H_ * (S_ / 32) * 256) return;
  int hg  = (int)(idx & 7);
  long r  = idx >> 3;
  int key = (int)(r & 31); r >>= 5;
  int kb  = (int)(r % (S_ / 32)); r /= (S_ / 32);
  int h   = (int)(r % H_);
  int b   = (int)(r / H_);
  int skey = kb * 32 + key;
  const float* src = qkv + ((size_t)(b * S_ + skey)) * 3 * DIM_ + DIM_ + h * HD_;
  const float* cr = cs + (size_t)skey * HD_;
  const float* sr = sn + (size_t)skey * HD_;
  v8bf v;
  for (int i = 0; i < 8; i++) {
    int hd = hg * 8 + i;
    float val = (hd < 32) ? src[hd] * cr[hd] - src[hd + 32] * sr[hd]
                          : src[hd] * cr[hd] + src[hd - 32] * sr[hd];
    v[i] = (__bf16)val;
  }
  size_t off = (((size_t)(b * H_ + h)) * (S_ / 32) + kb) * 2048
             + (size_t)((hg >> 2) * 1024 + (key >> 4) * 512
                        + ((key & 15) + 16 * (hg & 1)) * 16 + ((hg >> 1) & 1) * 8);
  *(v8bf*)(ksz + off) = v;
}

// V swizzle into V fragment tiles: per (b,h,keyblock) 2048 elems:
//   [sub4, stride 1024][lane32][j16]; lane=(hd&15)+16*(kg&1), j0=(kg>>1)*8
__global__ __launch_bounds__(256)
void v_swz_kernel(const float* __restrict__ qkv, __bf16* __restrict__ vsz) {
  long idx = (long)blockIdx.x * 256 + threadIdx.x;
  if (idx >= (long)B_ * H_ * (S_ / 32) * 256) return;
  int hd = (int)(idx & 63);
  long r = idx >> 6;
  int kg = (int)(r & 3); r >>= 2;
  int kb = (int)(r % (S_ / 32)); r /= (S_ / 32);
  int h  = (int)(r % H_);
  int b  = (int)(r / H_);
  v8bf v;
  for (int i = 0; i < 8; i++)
    v[i] = (__bf16)qkv[((size_t)(b * S_ + kb * 32 + kg * 8 + i)) * 3 * DIM_
                       + 2 * DIM_ + h * HD_ + hd];
  size_t off = (((size_t)(b * H_ + h)) * (S_ / 32) + kb) * 2048
             + (size_t)((hd >> 4) * 1024 + ((hd & 15) + 16 * (kg & 1)) * 16 + (kg >> 1) * 8);
  *(v8bf*)(vsz + off) = v;
}

// ------- WMMA GEMM, pre-swizzled bf16 operands, async double-buffered -------
// C = act(A @ B + bias) (+res); 128x64 tile, 8 waves, 1 barrier per K-step.
__global__ __launch_bounds__(256)
void wmma_gemm_async(const __bf16* __restrict__ Az, const __bf16* __restrict__ Bz,
                     const float* __restrict__ bias, const float* __restrict__ res,
                     float* __restrict__ C, int M, int N, int K,
                     long long sAe, long long sBe, long long sBias, long long sC,
                     int act) {
  __shared__ v16bf Asw[2][8][32];   // ping-pong 8KB A tiles
  __shared__ v16bf Bsw[2][4][32];   // ping-pong 4KB B tiles
  const int wave = threadIdx.x >> 5;
  const int lane = threadIdx.x & 31;
  const int hi = (lane >= 16), lm = lane & 15;
  const int z  = blockIdx.z;
  const int mt = blockIdx.y, nt = blockIdx.x;
  const int KT = K >> 5;
  const __bf16* Abase = Az + (size_t)z * sAe + (size_t)mt * KT * 4096;
  const __bf16* Bbase = Bz + (size_t)z * sBe + (size_t)nt * KT * 2048;
  const float* biasz = bias ? bias + (size_t)z * sBias : nullptr;
  const float* resz  = res  ? res  + (size_t)z * sC    : nullptr;
  float* Cz = C + (size_t)z * sC;

  unsigned ldsA[2] = { lds_off_of(&Asw[0][0][0]), lds_off_of(&Asw[1][0][0]) };
  unsigned ldsB[2] = { lds_off_of(&Bsw[0][0][0]), lds_off_of(&Bsw[1][0][0]) };
  const unsigned t16 = threadIdx.x * 16;   // 16B per unit

  // prologue: async-copy tile kt=0 into buffer 0
  async_b128(ldsA[0] + t16,        Abase + (size_t)threadIdx.x * 8);
  async_b128(ldsA[0] + t16 + 4096, Abase + 2048 + (size_t)threadIdx.x * 8);
  async_b128(ldsB[0] + t16,        Bbase + (size_t)threadIdx.x * 8);

  v8f acc[4];
  for (int n = 0; n < 4; n++) for (int r = 0; r < 8; r++) acc[n][r] = 0.0f;

  for (int kt = 0; kt < KT; kt++) {
    int p = kt & 1;
    wait_async0();      // own copies done (in-order per wave)
    __syncthreads();    // everyone's copies done; prior compute on p^1 done
    if (kt + 1 < KT) {  // overlap next tile's DMA with this tile's WMMAs
      const __bf16* an = Abase + (size_t)(kt + 1) * 4096 + (size_t)threadIdx.x * 8;
      const __bf16* bn = Bbase + (size_t)(kt + 1) * 2048 + (size_t)threadIdx.x * 8;
      async_b128(ldsA[p ^ 1] + t16,        an);
      async_b128(ldsA[p ^ 1] + t16 + 4096, an + 2048);
      async_b128(ldsB[p ^ 1] + t16,        bn);
    }
    v16bf a = Asw[p][wave][lane];
    for (int n = 0; n < 4; n++)
      acc[n] = wmma_bf16(a, Bsw[p][n][lane], acc[n]);
  }
  const int m0 = mt * 128, n0 = nt * 64;
  for (int n = 0; n < 4; n++) {
    const int gn = n0 + n * 16 + lm;
    const float bv = biasz ? biasz[gn] : 0.0f;
    for (int r = 0; r < 8; r++) {
      const int gm = m0 + wave * 16 + r + hi * 8;
      float v = acc[n][r] + bv;
      if (act) v = gelu_f(v);
      size_t ci = (size_t)gm * N + gn;
      if (resz) v += resz[ci];
      Cz[ci] = v;
    }
  }
}

// ------- flash attention: pre-swizzled K^T/V, async double-buffered ---------
__global__ __launch_bounds__(128)
void flash_attn_async(const float* __restrict__ qkv, const __bf16* __restrict__ ksz,
                      const __bf16* __restrict__ vsz, float* __restrict__ oat) {
  __shared__ v16bf Ksw[2][2][2][32];  // ping-pong 4KB K^T tiles
  __shared__ v16bf Vsw[2][4][32];     // ping-pong 4KB V tiles
  __shared__ v16bf Psw[4][32];
  __shared__ float sc[4][16][32];
  __shared__ float rowM[4][16], rowL[4][16], rowS[4][16];
  const int wave = threadIdx.x >> 5;
  const int lane = threadIdx.x & 31;
  const int hi   = (lane >= 16);
  const int lm   = lane & 15;
  const int b = blockIdx.z, h = blockIdx.y;
  const int qbase = blockIdx.x * 64 + wave * 16;

  const __bf16* kbase = ksz + (((size_t)(b * H_ + h)) * (S_ / 32)) * 2048;
  const __bf16* vbase = vsz + (((size_t)(b * H_ + h)) * (S_ / 32)) * 2048;
  unsigned ldsK[2] = { lds_off_of(&Ksw[0][0][0][0]), lds_off_of(&Ksw[1][0][0][0]) };
  unsigned ldsV[2] = { lds_off_of(&Vsw[0][0][0]),    lds_off_of(&Vsw[1][0][0]) };
  const unsigned t32 = threadIdx.x * 32;  // 32B (= 16 elems) per thread per tile

  v16bf aq0, aq1;  // Q fragment, pre-scaled by 1/sqrt(64)
  {
    size_t row = ((size_t)(b * S_ + qbase + lm)) * 3 * DIM_ + h * HD_;
    for (int g = 0; g < 2; g++)
      for (int i = 0; i < 8; i++) {
        int k = g * 16 + hi * 8 + i;
        aq0[g * 8 + i] = (__bf16)(qkv[row + k]      * 0.125f);
        aq1[g * 8 + i] = (__bf16)(qkv[row + 32 + k] * 0.125f);
      }
  }
  v8f o[4];
  for (int n = 0; n < 4; n++) for (int r = 0; r < 8; r++) o[n][r] = 0.0f;
  if (lane < 16) { rowM[wave][lane] = -3.0e38f; rowL[wave][lane] = 0.0f; }

  // prologue: async-copy key-block 0 into buffer 0
  {
    const __bf16* kn = kbase + (size_t)threadIdx.x * 16;
    const __bf16* vn = vbase + (size_t)threadIdx.x * 16;
    async_b128(ldsK[0] + t32,      kn);
    async_b128(ldsK[0] + t32 + 16, kn + 8);
    async_b128(ldsV[0] + t32,      vn);
    async_b128(ldsV[0] + t32 + 16, vn + 8);
  }

  for (int it = 0; it < S_ / 32; it++) {
    int p = it & 1;
    wait_async0();
    __syncthreads();
    if (it + 1 < S_ / 32) {
      const __bf16* kn = kbase + (size_t)(it + 1) * 2048 + (size_t)threadIdx.x * 16;
      const __bf16* vn = vbase + (size_t)(it + 1) * 2048 + (size_t)threadIdx.x * 16;
      async_b128(ldsK[p ^ 1] + t32,      kn);
      async_b128(ldsK[p ^ 1] + t32 + 16, kn + 8);
      async_b128(ldsV[p ^ 1] + t32,      vn);
      async_b128(ldsV[p ^ 1] + t32 + 16, vn + 8);
    }
    // ---- scores: S = Q K^T ----
    v8f s0, s1;
    for (int r = 0; r < 8; r++) { s0[r] = 0.f; s1[r] = 0.f; }
    s0 = wmma_bf16(aq0, Ksw[p][0][0][lane], s0);
    s0 = wmma_bf16(aq1, Ksw[p][1][0][lane], s0);
    s1 = wmma_bf16(aq0, Ksw[p][0][1][lane], s1);
    s1 = wmma_bf16(aq1, Ksw[p][1][1][lane], s1);
    for (int r = 0; r < 8; r++) {
      sc[wave][r + hi * 8][lm]      = s0[r];
      sc[wave][r + hi * 8][16 + lm] = s1[r];
    }
    __syncthreads();
    // ---- online softmax; lane m owns query row m ----
    if (lane < 16) {
      const int m = lane;
      float oldm = rowM[wave][m];
      float tm = oldm;
      for (int c = 0; c < 32; c++) tm = fmaxf(tm, sc[wave][m][c]);
      float scl = __expf(oldm - tm);
      float rs = 0.f;
      for (int c = 0; c < 32; c++) {
        float pv = __expf(sc[wave][m][c] - tm);
        rs += pv;
        ((__bf16*)&Psw[wave][m + 16 * ((c >> 3) & 1)])[(c & 7) + 8 * (c >> 4)] =
            (__bf16)pv;
      }
      rowM[wave][m] = tm;
      rowL[wave][m] = rowL[wave][m] * scl + rs;
      rowS[wave][m] = scl;
    }
    __syncthreads();
    // ---- rescale O, then O += P V ----
    for (int r = 0; r < 8; r++) {
      float f = rowS[wave][r + hi * 8];
      for (int n = 0; n < 4; n++) o[n][r] *= f;
    }
    v16bf pA = Psw[wave][lane];
    for (int n = 0; n < 4; n++)
      o[n] = wmma_bf16(pA, Vsw[p][n][lane], o[n]);
  }
  for (int n = 0; n < 4; n++) {
    for (int r = 0; r < 8; r++) {
      const int m = r + hi * 8;
      const float inv = 1.0f / rowL[wave][m];
      const int q = qbase + m;
      oat[((size_t)(b * S_ + q)) * DIM_ + h * HD_ + n * 16 + lm] = o[n][r] * inv;
    }
  }
}

// ---------------- MoE routing -----------------------------------------------
__global__ __launch_bounds__(256)
void gate_kernel(const float* __restrict__ h2, const float* __restrict__ gw,
                 int* __restrict__ topi, float* __restrict__ topw,
                 float* __restrict__ me, float* __restrict__ ce) {
  const int wave = threadIdx.x >> 5, lane = threadIdx.x & 31;
  const int t = blockIdx.x * 8 + wave;
  float acc[E_];
  for (int e = 0; e < E_; e++) acc[e] = 0.f;
  const float* xr = h2 + (size_t)t * DIM_;
  for (int d = lane; d < DIM_; d += 32) {
    float xv = xr[d];
    const float* g = gw + (size_t)d * E_;
    for (int e = 0; e < E_; e++) acc[e] += xv * g[e];
  }
  for (int e = 0; e < E_; e++)
    for (int off = 16; off > 0; off >>= 1)
      acc[e] += __shfl_xor(acc[e], off, 32);
  if (lane == 0) {
    float mx = acc[0];
    for (int e = 1; e < E_; e++) mx = fmaxf(mx, acc[e]);
    float p[E_], sum = 0.f;
    for (int e = 0; e < E_; e++) { p[e] = __expf(acc[e] - mx); sum += p[e]; }
    for (int e = 0; e < E_; e++) p[e] /= sum;
    int i1 = 0;
    for (int e = 1; e < E_; e++) if (p[e] > p[i1]) i1 = e;
    int i2 = -1;
    for (int e = 0; e < E_; e++) { if (e == i1) continue; if (i2 < 0 || p[e] > p[i2]) i2 = e; }
    float wsum = p[i1] + p[i2];
    topi[t * 2] = i1;  topi[t * 2 + 1] = i2;
    topw[t * 2] = p[i1] / wsum; topw[t * 2 + 1] = p[i2] / wsum;
    for (int e = 0; e < E_; e++) atomicAdd(&me[e], p[e]);
    atomicAdd(&ce[i1], 1.f); atomicAdd(&ce[i2], 1.f);
  }
}

__global__ void pos_kernel(const int* __restrict__ topi, int* __restrict__ pos) {
  const int e = threadIdx.x;
  if (e >= E_) return;
  int cnt = 0;
  for (int i = 0; i < 2 * T_; i++) {
    int t = i & (T_ - 1);
    int k = i >> 12;  // T_ == 4096
    if (topi[t * 2 + k] == e) { pos[i] = cnt; cnt++; }
  }
}

__global__ __launch_bounds__(256)
void scatter_kernel(const float* __restrict__ h2, const int* __restrict__ topi,
                    const int* __restrict__ pos, float* __restrict__ buf) {
  const int i = blockIdx.x;
  const int p = pos[i];
  if (p >= CAP_) return;
  const int t = i & (T_ - 1), k = i >> 12;
  const int e = topi[t * 2 + k];
  const float* src = h2 + (size_t)t * DIM_;
  float* dst = buf + ((size_t)e * CAP_ + p) * DIM_;
  for (int d = threadIdx.x; d < DIM_; d += 256) dst[d] = src[d];
}

__global__ __launch_bounds__(256)
void finalize_kernel(const float* __restrict__ xat, const float* __restrict__ oute,
                     const int* __restrict__ topi, const int* __restrict__ pos,
                     const float* __restrict__ topw, float* __restrict__ out) {
  const long idx = (long)blockIdx.x * 256 + threadIdx.x;
  const int d = idx & (DIM_ - 1);
  const int t = (int)(idx >> 10);
  float y = 0.f;
  for (int k = 0; k < 2; k++) {
    int p = pos[k * T_ + t];
    if (p < CAP_) {
      int e = topi[t * 2 + k];
      y += topw[t * 2 + k] * oute[((size_t)e * CAP_ + p) * DIM_ + d];
    }
  }
  out[idx] = xat[idx] + y;
}

__global__ void aux_kernel(const float* __restrict__ me, const float* __restrict__ ce,
                           float* __restrict__ out) {
  if (threadIdx.x == 0) {
    float s = 0.f;
    for (int e = 0; e < E_; e++)
      s += (me[e] / (float)T_) * (ce[e] / ((float)T_ * 2.0f));
    out[0] = (float)E_ * s;
  }
}

// ---------------- host orchestration ----------------------------------------
extern "C" void kernel_launch(void* const* d_in, const int* in_sizes, int n_in,
                              void* d_out, int out_size, void* d_ws, size_t ws_size,
                              hipStream_t stream) {
  (void)in_sizes; (void)n_in; (void)out_size; (void)ws_size;
  const float* x      = (const float*)d_in[0];
  const float* cosb   = (const float*)d_in[1];
  const float* sinb   = (const float*)d_in[2];
  const float* ln1_g  = (const float*)d_in[3];
  const float* ln1_b  = (const float*)d_in[4];
  const float* ln2_g  = (const float*)d_in[5];
  const float* ln2_b  = (const float*)d_in[6];
  const float* wqkv   = (const float*)d_in[7];
  const float* bqkv   = (const float*)d_in[8];
  const float* wo     = (const float*)d_in[9];
  const float* bo     = (const float*)d_in[10];
  const float* gate_w = (const float*)d_in[11];
  const float* w1     = (const float*)d_in[12];
  const float* b1     = (const float*)d_in[13];
  const float* w2     = (const float*)d_in[14];
  const float* b2     = (const float*)d_in[15];
  float* out = (float*)d_out;

  char* wp = (char*)d_ws;
  auto allocf = [&](size_t n) -> float* {
    float* r = (float*)wp; wp += n * sizeof(float); return r;
  };
  auto allocb = [&](size_t n) -> __bf16* {
    __bf16* r = (__bf16*)wp; wp += n * sizeof(__bf16); return r;
  };
  float*  h      = allocf((size_t)T_ * DIM_);
  float*  qkv    = allocf((size_t)T_ * 3 * DIM_);
  float*  oat    = allocf((size_t)T_ * DIM_);
  float*  xat    = allocf((size_t)T_ * DIM_);
  float*  h2     = allocf((size_t)T_ * DIM_);
  float*  buf    = allocf((size_t)E_ * CAP_ * DIM_);
  float*  hh     = allocf((size_t)E_ * CAP_ * HID_);
  float*  oute   = allocf((size_t)E_ * CAP_ * DIM_);
  __bf16* hsz    = allocb((size_t)T_ * DIM_);
  __bf16* oasz   = allocb((size_t)T_ * DIM_);
  __bf16* bufsz  = allocb((size_t)E_ * CAP_ * DIM_);
  __bf16* hhsz   = allocb((size_t)E_ * CAP_ * HID_);
  __bf16* wqkvsz = allocb((size_t)DIM_ * 3 * DIM_);
  __bf16* wosz   = allocb((size_t)DIM_ * DIM_);
  __bf16* w1sz   = allocb((size_t)E_ * DIM_ * HID_);
  __bf16* w2sz   = allocb((size_t)E_ * HID_ * DIM_);
  __bf16* ksz    = allocb((size_t)B_ * H_ * (S_ / 32) * 2048);
  __bf16* vsz    = allocb((size_t)B_ * H_ * (S_ / 32) * 2048);
  int*    topi   = (int*)allocf(2 * T_);
  float*  topw   = allocf(2 * T_);
  int*    pos    = (int*)allocf(2 * T_);
  float*  me     = allocf(E_);
  float*  ce     = allocf(E_);

  auto gridFor = [](long total) { return (unsigned)((total + 255) / 256); };

  // 0) one-time weight swizzles (bf16 fragment tiles)
  { long n = (long)(DIM_ / 8) * 3 * DIM_;
    swizzleB_kernel<<<gridFor(n), 256, 0, stream>>>(wqkv, wqkvsz, DIM_, 3 * DIM_, 0, n); }
  { long n = (long)(DIM_ / 8) * DIM_;
    swizzleB_kernel<<<gridFor(n), 256, 0, stream>>>(wo, wosz, DIM_, DIM_, 0, n); }
  { long n = (long)E_ * (DIM_ / 8) * HID_;
    swizzleB_kernel<<<gridFor(n), 256, 0, stream>>>(w1, w1sz, DIM_, HID_,
                                                    (long long)DIM_ * HID_, n); }
  { long n = (long)E_ * (HID_ / 8) * DIM_;
    swizzleB_kernel<<<gridFor(n), 256, 0, stream>>>(w2, w2sz, HID_, DIM_,
                                                    (long long)HID_ * DIM_, n); }

  // 1) h = LN1(x); swizzle for GEMM A
  layernorm_kernel<<<T_, 256, 0, stream>>>(x, ln1_g, ln1_b, h);
  { long n = (long)T_ * (DIM_ / 8);
    swizzleA_kernel<<<gridFor(n), 256, 0, stream>>>(h, hsz, T_, DIM_, 0, n); }

  // 2) qkv = h @ wqkv + bqkv
  wmma_gemm_async<<<dim3(3 * DIM_ / 64, T_ / 128, 1), 256, 0, stream>>>(
      hsz, wqkvsz, bqkv, nullptr, qkv, T_, 3 * DIM_, DIM_, 0, 0, 0, 0, 0);

  // 3) RoPE: Q in place; K fused into swizzle; V swizzle
  { long n = (long)T_ * H_ * 32;
    rope_q_kernel<<<gridFor(n), 256, 0, stream>>>(qkv, cosb, sinb); }
  { long n = (long)B_ * H_ * (S_ / 32) * 256;
    ropek_swz_kernel<<<gridFor(n), 256, 0, stream>>>(qkv, cosb, sinb, ksz);
    v_swz_kernel<<<gridFor(n), 256, 0, stream>>>(qkv, vsz); }

  // 4) flash attention -> oat
  flash_attn_async<<<dim3(S_ / 64, H_, B_), 128, 0, stream>>>(qkv, ksz, vsz, oat);

  // 5) xat = x + oat @ wo + bo
  { long n = (long)T_ * (DIM_ / 8);
    swizzleA_kernel<<<gridFor(n), 256, 0, stream>>>(oat, oasz, T_, DIM_, 0, n); }
  wmma_gemm_async<<<dim3(DIM_ / 64, T_ / 128, 1), 256, 0, stream>>>(
      oasz, wosz, bo, x, xat, T_, DIM_, DIM_, 0, 0, 0, 0, 0);

  // 6) h2 = LN2(xat)
  layernorm_kernel<<<T_, 256, 0, stream>>>(xat, ln2_g, ln2_b, h2);

  // 7) routing
  zero_kernel<<<1, 32, 0, stream>>>(me, 2 * E_);
  { long n = (long)E_ * CAP_ * DIM_;
    zero_kernel<<<gridFor(n), 256, 0, stream>>>(buf, n); }
  gate_kernel<<<T_ / 8, 256, 0, stream>>>(h2, gate_w, topi, topw, me, ce);
  pos_kernel<<<1, E_, 0, stream>>>(topi, pos);
  scatter_kernel<<<2 * T_, 256, 0, stream>>>(h2, topi, pos, buf);

  // 8) expert FFN (batched over 8 experts)
  { long n = (long)E_ * CAP_ * (DIM_ / 8);
    swizzleA_kernel<<<gridFor(n), 256, 0, stream>>>(buf, bufsz, CAP_, DIM_,
                                                    (long long)CAP_ * DIM_, n); }
  wmma_gemm_async<<<dim3(HID_ / 64, CAP_ / 128, E_), 256, 0, stream>>>(
      bufsz, w1sz, b1, nullptr, hh, CAP_, HID_, DIM_,
      (long long)CAP_ * DIM_, (long long)DIM_ * HID_, HID_, (long long)CAP_ * HID_, 1);
  { long n = (long)E_ * CAP_ * (HID_ / 8);
    swizzleA_kernel<<<gridFor(n), 256, 0, stream>>>(hh, hhsz, CAP_, HID_,
                                                    (long long)CAP_ * HID_, n); }
  wmma_gemm_async<<<dim3(DIM_ / 64, CAP_ / 128, E_), 256, 0, stream>>>(
      hhsz, w2sz, b2, nullptr, oute, CAP_, DIM_, HID_,
      (long long)CAP_ * HID_, (long long)HID_ * DIM_, DIM_, (long long)CAP_ * DIM_, 0);

  // 9) combine + aux
  finalize_kernel<<<(T_ * DIM_) / 256, 256, 0, stream>>>(xat, oute, topi, pos, topw, out);
  aux_kernel<<<1, 1, 0, stream>>>(me, ce, out + (size_t)T_ * DIM_);
}
